// MuCalculator_46737834115727
// MI455X (gfx1250) — matrix operand order repaired and checked by reading, not compiled, not run on hardware
//
#include <hip/hip_runtime.h>
#include <hip/hip_bf16.h>

typedef __attribute__((ext_vector_type(16))) _Float16 v16h;
typedef __attribute__((ext_vector_type(8)))  float    v8f;

#define K_NB  32
#define H_DIM 1024

// ---------------------------------------------------------------------------
// Kernel 1: context distances. Memory-bound (streams 1 GiB of hiddens).
// One block per batch row; cache_hidden[b] staged once in LDS so it is read
// from HBM exactly once (not 32x). 8 waves x 4 neighbors each; wave32
// shuffle reduction over 1024 elements (32 lanes x 8 float4).
// ---------------------------------------------------------------------------
__global__ __launch_bounds__(256) void ctxdist_kernel(
    const float* __restrict__ cache_hidden,
    const float* __restrict__ hiddens,
    float* __restrict__ ctx_raw)
{
  __shared__ float csm[H_DIM];
  const int b    = blockIdx.x;
  const int tid  = threadIdx.x;
  const int wave = tid >> 5;
  const int lane = tid & 31;

  const float4* cg = reinterpret_cast<const float4*>(cache_hidden + (size_t)b * H_DIM);
  float4* cs = reinterpret_cast<float4*>(csm);
  cs[tid] = cg[tid];                     // 256 threads * float4 = 1024 floats
  __syncthreads();

  for (int jj = 0; jj < 4; ++jj) {
    const int j = wave * 4 + jj;
    const float4* hg = reinterpret_cast<const float4*>(
        hiddens + ((size_t)b * K_NB + j) * H_DIM);
    float s = 0.0f;
#pragma unroll
    for (int i = 0; i < 8; ++i) {
      float4 hv = hg[i * 32 + lane];     // coalesced 512B per wave step
      float4 cv = cs[i * 32 + lane];     // conflict-free 16B LDS reads
      float dx = cv.x - hv.x, dy = cv.y - hv.y;
      float dz = cv.z - hv.z, dw = cv.w - hv.w;
      s = fmaf(dx, dx, s); s = fmaf(dy, dy, s);
      s = fmaf(dz, dz, s); s = fmaf(dw, dw, s);
    }
#pragma unroll
    for (int off = 16; off; off >>= 1) s += __shfl_xor(s, off, 32);
    if (lane == 0) ctx_raw[(size_t)b * K_NB + j] = sqrtf(s);
  }
}

// ---------------------------------------------------------------------------
// Kernel 2: prefix-distinct label counts + MetaK MLP via WMMA + outputs.
// One wave32 per 16 batch rows. Layer 1 (x[16,96] @ W1^T[96,64]) uses
// v_wmma_f32_16x16x32_f16: 3 K-chunks x 4 N-tiles = 12 WMMAs per wave.
// ---------------------------------------------------------------------------
__global__ __launch_bounds__(32) void metak_mlp_kernel(
    const int*   __restrict__ vals,
    const float* __restrict__ dist,
    const float* __restrict__ ctx_raw,
    const float* __restrict__ W1,     // [64][96] row-major
    const float* __restrict__ b1,     // [64]
    const float* __restrict__ W2,     // [64]
    const float* __restrict__ b2,     // [1]
    float* __restrict__ out,
    size_t half_off)                  // = B*K, offset of second output
{
  __shared__ float xs[16][96];        // [ctx | dist | counts] per row (f32)
  __shared__ int   vsm[16][K_NB];
  __shared__ float hidls[16][64];
  __shared__ float scl[16];

  const int lane = threadIdx.x;       // 0..31 (wave32)
  const int b0   = blockIdx.x * 16;
  const int hi   = lane >> 4;         // half-wave select
  const int nl   = lane & 15;

  // Stage per-row inputs (lane == neighbor index j).
  for (int m = 0; m < 16; ++m) {
    const size_t base = (size_t)(b0 + m) * K_NB;
    xs[m][lane]        = ctx_raw[base + lane];
    xs[m][32 + lane]   = dist[base + lane];
    vsm[m][lane]       = vals[base + lane];
  }
  __syncthreads();

  // Prefix count of distinct non-zero labels (lanes 0..15, one row each).
  // counts[j] = #{ i<=j : v[i]!=0 and v[i] not seen before position i }.
  if (lane < 16) {
    int cnt = 0;
    for (int i = 0; i < K_NB; ++i) {
      const int v = vsm[lane][i];
      int first = (v != 0) ? 1 : 0;
      for (int p = 0; p < i; ++p)
        if (vsm[lane][p] == v) first = 0;
      cnt += first;
      xs[lane][64 + i] = (float)cnt;
    }
  }
  __syncthreads();

  // ---- Layer 1 via WMMA f16 -> f32: D[16x16] += A[16x32] * B[32x16] ----
  v8f acc[4];
#pragma unroll
  for (int nt = 0; nt < 4; ++nt)
#pragma unroll
    for (int e = 0; e < 8; ++e) acc[nt][e] = 0.0f;

  for (int kc = 0; kc < 3; ++kc) {
    // A fragment (16-bit A 16x32 layout): row M = nl for all lanes;
    // VGPR g, elem t: K = kc*32 + 16*(g>>2) + 2*(g&3) + t + 8*hi
    v16h a;
#pragma unroll
    for (int g = 0; g < 8; ++g)
#pragma unroll
      for (int t = 0; t < 2; ++t) {
        const int kk = kc * 32 + 16 * (g >> 2) + 2 * (g & 3) + t + 8 * hi;
        a[2 * g + t] = (_Float16)xs[nl][kk];
      }
#pragma unroll
    for (int nt = 0; nt < 4; ++nt) {
      // B fragment (32x16): same K order as A: K = 16*(g>>2) + 2*(g&3) + t + 8*hi;
      // column N = nt*16 + nl. B[k][n] = W1[n][k]  (x @ W1^T).
      v16h bf;
#pragma unroll
      for (int g = 0; g < 8; ++g)
#pragma unroll
        for (int t = 0; t < 2; ++t) {
          const int kk = kc * 32 + 16 * (g >> 2) + 2 * (g & 3) + t + 8 * hi;
          bf[2 * g + t] = (_Float16)W1[(size_t)(nt * 16 + nl) * 96 + kk];
        }
      acc[nt] = __builtin_amdgcn_wmma_f32_16x16x32_f16(
          false, a, false, bf, (short)0, acc[nt], false, false);
    }
  }

  // Bias + tanh; C/D layout: VGPR r -> M = r + 8*hi, N = nt*16 + nl.
#pragma unroll
  for (int nt = 0; nt < 4; ++nt)
#pragma unroll
    for (int r = 0; r < 8; ++r) {
      const int m = r + 8 * hi;
      const int n = nt * 16 + nl;
      hidls[m][n] = tanhf(acc[nt][r] + b1[n]);
    }
  __syncthreads();

  // ---- Layer 2 (64 -> 1) + sigmoid, trivially small -> VALU ----
  if (lane < 16) {
    float s = b2[0];
    for (int n = 0; n < 64; ++n) s = fmaf(hidls[lane][n], W2[n], s);
    scl[lane] = 5.0f / (1.0f + expf(-s));   // 5 * sigmoid(s)
  }
  __syncthreads();

  // ---- Outputs: (ctx*scale, dist + ctx*scale) ----
  for (int m = 0; m < 16; ++m) {
    const size_t base = (size_t)(b0 + m) * K_NB;
    const float cd = xs[m][lane] * scl[m];
    out[base + lane]            = cd;
    out[half_off + base + lane] = xs[m][32 + lane] + cd;
  }
}

extern "C" void kernel_launch(void* const* d_in, const int* in_sizes, int n_in,
                              void* d_out, int out_size, void* d_ws, size_t ws_size,
                              hipStream_t stream) {
  const int*   vals         = (const int*)  d_in[0];
  const float* distances    = (const float*)d_in[1];
  const float* cache_hidden = (const float*)d_in[2];
  const float* hiddens      = (const float*)d_in[3];
  const float* W1           = (const float*)d_in[4];
  const float* b1           = (const float*)d_in[5];
  const float* W2           = (const float*)d_in[6];
  const float* b2           = (const float*)d_in[7];
  float* out = (float*)d_out;

  const int B = in_sizes[0] / K_NB;           // vals is [B,1,K]
  float* ctx_raw = (float*)d_ws;              // B*K floats = 1 MiB scratch

  ctxdist_kernel<<<B, 256, 0, stream>>>(cache_hidden, hiddens, ctx_raw);

  metak_mlp_kernel<<<B / 16, 32, 0, stream>>>(
      vals, distances, ctx_raw, W1, b1, W2, b2, out, (size_t)B * K_NB);
}
